// PseudoQuantumFractalAttention_75960791597091
// MI455X (gfx1250) — compile-verified, matrix operand order checked by reading, not run on hardware
//
#include <hip/hip_runtime.h>

// ======================================================================
// PseudoQuantumFractalAttention on MI455X (gfx1250)
// Strategy: decompose all complex linears / attention into real bf16
// GEMMs executed by a tiled v_wmma_f32_16x16x32_bf16 kernel (f32 accum),
// plus small elementwise kernels for softmax(|.|), layernorm(|.|), LSTM.
// Compute-bound (~700 GFLOP, ~450 flop/byte) -> matrix cores, not HBM.
// GEMM uses double-buffered LDS: global loads of tile k+1 overlap the
// WMMAs of tile k; one workgroup barrier per 32-wide K step.
// ======================================================================

#define DEVFN __device__ __forceinline__

typedef __attribute__((ext_vector_type(16))) __bf16 v16bf;
typedef __attribute__((ext_vector_type(8)))  float  v8f;

union FragBF { v16bf v; uint4 q[2]; };

DEVFN unsigned short f2bf(float f) {
  unsigned int u = __float_as_uint(f);
  u += 0x7FFFu + ((u >> 16) & 1u);          // round-to-nearest-even
  return (unsigned short)(u >> 16);
}

DEVFN float sigmf(float x) { return 1.f / (1.f + __expf(-x)); }

// ---------------------------------------------------------------- convert
__global__ void k_f32_to_bf16(const float* __restrict__ src,
                              unsigned short* __restrict__ dst, size_t n) {
  size_t i = (size_t)blockIdx.x * blockDim.x + threadIdx.x;
  size_t stride = (size_t)gridDim.x * blockDim.x;
  for (; i < n; i += stride) dst[i] = f2bf(src[i]);
}

// ------------------------------------------------------------------ GEMM
// C[M,N] = alpha * A[M,K] @ B[N,K]^T  (+ C  if accumulate, else + bias)
// bias[n] = bias0[n] + bsign*bias1[n]  (either may be null)
// All of M%128==0, N%128==0, K%32==0 hold for every call in this model.
#define BM   128
#define BN   128
#define BKT  32
#define LDSS 40   // padded halves per LDS row (spreads banks)

__global__ __launch_bounds__(256)
void gemm_bf16_wmma(const unsigned short* __restrict__ A, int lda,
                    const unsigned short* __restrict__ Bm, int ldb,
                    float* __restrict__ C, int ldc,
                    int M, int N, int K,
                    float alpha, int accumulate,
                    const float* __restrict__ bias0,
                    const float* __restrict__ bias1, float bsign)
{
  __shared__ unsigned short ldsA[2][BM * LDSS];
  __shared__ unsigned short ldsB[2][BN * LDSS];

  const int tid   = threadIdx.x;
  const int lane  = tid & 31;
  const int wave  = tid >> 5;                 // 8 wave32s
  const int wm0   = (wave >> 1) * 32;         // wave tile: 32 rows
  const int wn0   = (wave & 1)  * 64;         // wave tile: 64 cols
  const int row16 = lane & 15;
  const int hi    = lane >> 4;                // half-wave select (K phase)

  const int tileM = blockIdx.y * BM;
  const int tileN = blockIdx.x * BN;

  // cooperative tile load mapping: 256 threads, 2 per row, 16 halves each
  const int lr = tid >> 1;
  const int lc = (tid & 1) << 4;
  const unsigned short* gA = A  + (size_t)(tileM + lr) * lda + lc;
  const unsigned short* gB = Bm + (size_t)(tileN + lr) * ldb + lc;

  v8f acc[2][4] = {};

  const int nIter = K / BKT;

  // ---- prologue: stage tile 0 into LDS buffer 0 ----
  {
    const uint4* pa = (const uint4*)gA;
    const uint4* pb = (const uint4*)gB;
    uint4 a0 = pa[0], a1 = pa[1];
    uint4 b0 = pb[0], b1 = pb[1];
    uint4* la = (uint4*)&ldsA[0][lr * LDSS + lc];
    uint4* lb = (uint4*)&ldsB[0][lr * LDSS + lc];
    la[0] = a0; la[1] = a1;
    lb[0] = b0; lb[1] = b1;
  }
  __syncthreads();

  for (int it = 0; it < nIter; ++it) {
    const int buf = it & 1;

    // ---- issue global loads for tile it+1 (overlaps WMMAs below) ----
    uint4 a0, a1, b0, b1;
    const bool more = (it + 1) < nIter;
    if (more) {
      const int k1 = (it + 1) * BKT;
      const uint4* pa = (const uint4*)(gA + k1);
      const uint4* pb = (const uint4*)(gB + k1);
      a0 = pa[0]; a1 = pa[1];
      b0 = pb[0]; b1 = pb[1];
      if (it + 2 < nIter) {                  // gfx1250 global_prefetch_b8
        __builtin_prefetch(gA + k1 + BKT, 0, 1);
        __builtin_prefetch(gB + k1 + BKT, 0, 1);
      }
    }

    // ---- fragments per the 16-bit operand layout:
    // lanes 0-15 hold K {0..7,16..23}, lanes 16-31 hold K {8..15,24..31}
    FragBF fa[2], fb[4];
    #pragma unroll
    for (int i = 0; i < 2; ++i) {
      const unsigned short* p = &ldsA[buf][(wm0 + i * 16 + row16) * LDSS + hi * 8];
      fa[i].q[0] = *(const uint4*)p;
      fa[i].q[1] = *(const uint4*)(p + 16);
    }
    #pragma unroll
    for (int j = 0; j < 4; ++j) {
      const unsigned short* p = &ldsB[buf][(wn0 + j * 16 + row16) * LDSS + hi * 8];
      fb[j].q[0] = *(const uint4*)p;
      fb[j].q[1] = *(const uint4*)(p + 16);
    }
    #pragma unroll
    for (int i = 0; i < 2; ++i)
      #pragma unroll
      for (int j = 0; j < 4; ++j)
        acc[i][j] = __builtin_amdgcn_wmma_f32_16x16x32_bf16(
            false, fa[i].v, false, fb[j].v, (short)0, acc[i][j], false, false);

    // ---- stage tile it+1 into the other buffer; one barrier per step ----
    if (more) {
      uint4* la = (uint4*)&ldsA[buf ^ 1][lr * LDSS + lc];
      uint4* lb = (uint4*)&ldsB[buf ^ 1][lr * LDSS + lc];
      la[0] = a0; la[1] = a1;
      lb[0] = b0; lb[1] = b1;
      __syncthreads();
    }
  }

  // C/D layout: VGPR r, lanes 0-15 -> M=r, lanes 16-31 -> M=r+8; N=lane&15
  #pragma unroll
  for (int i = 0; i < 2; ++i) {
    #pragma unroll
    for (int j = 0; j < 4; ++j) {
      const int n  = tileN + wn0 + j * 16 + row16;
      const int m0 = tileM + wm0 + i * 16 + hi * 8;
      float badd = 0.f;
      if (!accumulate) {
        if (bias0) badd = bias0[n];
        if (bias1) badd += bsign * bias1[n];
      }
      #pragma unroll
      for (int r = 0; r < 8; ++r) {
        size_t idx = (size_t)(m0 + r) * ldc + n;
        float v = alpha * acc[i][j][r];
        v += accumulate ? C[idx] : badd;
        C[idx] = v;
      }
    }
  }
}

// ------------------------------------------------- transpose f32 -> bf16
// Per-z matrix: src [rows][cols] f32 -> dst [cols][rows] bf16
__global__ void k_transpose_f32_to_bf16(const float* __restrict__ src,
                                        unsigned short* __restrict__ dst,
                                        int rows, int cols) {
  __shared__ float t[32][33];
  const size_t off = (size_t)blockIdx.z * rows * cols;
  src += off; dst += off;
  const int x  = blockIdx.x * 32 + threadIdx.x;   // source col
  const int y0 = blockIdx.y * 32;                 // source row base
  #pragma unroll
  for (int k = 0; k < 32; k += 8)
    t[threadIdx.y + k][threadIdx.x] = src[(size_t)(y0 + threadIdx.y + k) * cols + x];
  __syncthreads();
  const int oy0 = blockIdx.x * 32;                // dest row base (= src col)
  #pragma unroll
  for (int k = 0; k < 32; k += 8)
    dst[(size_t)(oy0 + threadIdx.y + k) * rows + (y0 + threadIdx.x)] =
        f2bf(t[threadIdx.x][threadIdx.y + k]);
}

// -------------------------------------------- softmax of |complex| rows
__global__ __launch_bounds__(256)
void k_abs_softmax(const float* __restrict__ re, const float* __restrict__ im,
                   unsigned short* __restrict__ attn, int cols, float scale) {
  __shared__ float red[256];
  const size_t base = (size_t)blockIdx.x * cols;
  const int tid = threadIdx.x;
  const int per = cols >> 8;                  // cols/256 == 4 here
  float av[4];
  float lmax = -3.4e38f;
  for (int k = 0; k < per; ++k) {
    int c = tid + (k << 8);
    float r = re[base + c], m = im[base + c];
    float a = scale * sqrtf(r * r + m * m);
    av[k] = a;
    lmax = fmaxf(lmax, a);
  }
  red[tid] = lmax; __syncthreads();
  for (int s = 128; s > 0; s >>= 1) {
    if (tid < s) red[tid] = fmaxf(red[tid], red[tid + s]);
    __syncthreads();
  }
  float mx = red[0]; __syncthreads();
  float lsum = 0.f;
  for (int k = 0; k < per; ++k) { av[k] = __expf(av[k] - mx); lsum += av[k]; }
  red[tid] = lsum; __syncthreads();
  for (int s = 128; s > 0; s >>= 1) {
    if (tid < s) red[tid] += red[tid + s];
    __syncthreads();
  }
  float inv = 1.f / red[0];
  for (int k = 0; k < per; ++k)
    attn[base + tid + (k << 8)] = f2bf(av[k] * inv);
}

// -------------------------------------------- layernorm of |complex| rows
__global__ __launch_bounds__(256)
void k_abs_layernorm(const float* __restrict__ yre, const float* __restrict__ yim,
                     const float* __restrict__ g, const float* __restrict__ bb,
                     float* __restrict__ nf, unsigned short* __restrict__ nbf,
                     int cols) {
  __shared__ float red[256];
  const size_t base = (size_t)blockIdx.x * cols;
  const int tid = threadIdx.x;
  const int per = cols >> 8;                  // cols/256 == 2 here
  float av[2];
  float s = 0.f, s2 = 0.f;
  for (int k = 0; k < per; ++k) {
    int c = tid + (k << 8);
    float r = yre[base + c], m = yim[base + c];
    float a = sqrtf(r * r + m * m);
    av[k] = a; s += a; s2 += a * a;
  }
  red[tid] = s; __syncthreads();
  for (int q = 128; q > 0; q >>= 1) { if (tid < q) red[tid] += red[tid + q]; __syncthreads(); }
  float mu = red[0] / cols; __syncthreads();
  red[tid] = s2; __syncthreads();
  for (int q = 128; q > 0; q >>= 1) { if (tid < q) red[tid] += red[tid + q]; __syncthreads(); }
  float var  = red[0] / cols - mu * mu;
  float rstd = rsqrtf(var + 1e-5f);
  for (int k = 0; k < per; ++k) {
    int c = tid + (k << 8);
    float v = (av[k] - mu) * rstd * g[c] + bb[c];
    nf[base + c]  = v;
    nbf[base + c] = f2bf(v);
  }
}

// ----------------------------------------------------- LSTM gate + output
__global__ void k_lstm_out(const float* __restrict__ gates,
                           const float* __restrict__ nf,
                           float* __restrict__ out, int O, size_t total) {
  size_t i = (size_t)blockIdx.x * blockDim.x + threadIdx.x;
  size_t stride = (size_t)gridDim.x * blockDim.x;
  for (; i < total; i += stride) {
    size_t m = i / O; int o = (int)(i - m * O);
    const float* gr = gates + m * (size_t)(4 * O);
    float ig = gr[o], gg = gr[2 * O + o], og = gr[3 * O + o]; // f*c0 == 0
    float c  = sigmf(ig) * tanhf(gg);
    float hn = sigmf(og) * tanhf(c);
    out[i] = nf[i] + hn;
  }
}

// ======================================================================
extern "C" void kernel_launch(void* const* d_in, const int* in_sizes, int n_in,
                              void* d_out, int out_size, void* d_ws, size_t ws_size,
                              hipStream_t stream) {
  (void)in_sizes; (void)n_in; (void)out_size; (void)ws_size;
  constexpr int Bc = 2, Sc = 1024, Dc = 1024, Hc = 8, Ec = 1024, Oc = 512;
  constexpr int Mc = Bc * Sc;                    // 2048 token rows

  const float* x       = (const float*)d_in[0];
  const float* enc_Wr  = (const float*)d_in[1];
  const float* enc_Wi  = (const float*)d_in[2];
  const float* enc_br  = (const float*)d_in[3];
  const float* enc_bi  = (const float*)d_in[4];
  const float* pWr[3]  = { (const float*)d_in[5],  (const float*)d_in[9],  (const float*)d_in[13] };
  const float* pWi[3]  = { (const float*)d_in[6],  (const float*)d_in[10], (const float*)d_in[14] };
  const float* pbr[3]  = { (const float*)d_in[7],  (const float*)d_in[11], (const float*)d_in[15] };
  const float* pbi[3]  = { (const float*)d_in[8],  (const float*)d_in[12], (const float*)d_in[16] };
  const float* o_Wr    = (const float*)d_in[17];
  const float* o_Wi    = (const float*)d_in[18];
  const float* o_br    = (const float*)d_in[19];
  const float* o_bi    = (const float*)d_in[20];
  const float* ln_g    = (const float*)d_in[21];
  const float* ln_b    = (const float*)d_in[22];
  const float* lstmWih = (const float*)d_in[23];
  const float* lstmbih = (const float*)d_in[25];
  const float* lstmbhh = (const float*)d_in[26];
  float* out = (float*)d_out;

  // ---- workspace layout (cursor; ~1.2 GB total) ----
  char* cur = (char*)d_ws;
  auto alloc_bf = [&](size_t n) {
    unsigned short* p = (unsigned short*)cur;
    cur += ((n * 2 + 255) & ~(size_t)255); return p;
  };
  auto alloc_f = [&](size_t n) {
    float* p = (float*)cur;
    cur += ((n * 4 + 255) & ~(size_t)255); return p;
  };

  unsigned short* xbf      = alloc_bf((size_t)Mc * Dc);
  unsigned short* encWr_bf = alloc_bf((size_t)Ec * Dc);
  unsigned short* encWi_bf = alloc_bf((size_t)Ec * Dc);
  unsigned short *pWr_bf[3], *pWi_bf[3];
  for (int p = 0; p < 3; ++p) {
    pWr_bf[p] = alloc_bf((size_t)Hc * Ec * Ec);
    pWi_bf[p] = alloc_bf((size_t)Hc * Ec * Ec);
  }
  unsigned short* oWr_bf = alloc_bf((size_t)Oc * Hc * Ec);
  unsigned short* oWi_bf = alloc_bf((size_t)Oc * Hc * Ec);
  unsigned short* Wih_bf = alloc_bf((size_t)4 * Oc * Oc);

  float* h_re = alloc_f((size_t)Mc * Ec);
  float* h_im = alloc_f((size_t)Mc * Ec);
  unsigned short* h_re_bf = alloc_bf((size_t)Mc * Ec);
  unsigned short* h_im_bf = alloc_bf((size_t)Mc * Ec);

  const size_t QSZ = (size_t)Hc * Mc * Ec;        // 16M elems
  float *qre = alloc_f(QSZ), *qim = alloc_f(QSZ);
  float *kre = alloc_f(QSZ), *kim = alloc_f(QSZ);
  float *vre = alloc_f(QSZ), *vim = alloc_f(QSZ);
  unsigned short* qre_bf = alloc_bf(QSZ);
  unsigned short* qim_bf = alloc_bf(QSZ);
  unsigned short* kre_bf = alloc_bf(QSZ);
  unsigned short* kim_bf = alloc_bf(QSZ);
  unsigned short* vtre_bf = alloc_bf(QSZ);        // [h][b][E][S]
  unsigned short* vtim_bf = alloc_bf(QSZ);

  const size_t DSZ = (size_t)Hc * Bc * Sc * Sc;   // 16M
  float* dots_re = alloc_f(DSZ);
  float* dots_im = alloc_f(DSZ);
  unsigned short* attn_bf = alloc_bf(DSZ);

  float* out_re = alloc_f(QSZ);
  float* out_im = alloc_f(QSZ);
  unsigned short* out_re_bf = alloc_bf(QSZ);
  unsigned short* out_im_bf = alloc_bf(QSZ);

  float* y_re = alloc_f((size_t)Mc * Oc);
  float* y_im = alloc_f((size_t)Mc * Oc);
  float* n_f  = alloc_f((size_t)Mc * Oc);
  unsigned short* n_bf = alloc_bf((size_t)Mc * Oc);
  float* gates = alloc_f((size_t)Mc * 4 * Oc);

  // ---- launch helpers ----
  auto cvt = [&](const float* s, unsigned short* dptr, size_t n) {
    size_t blk = (n + 255) / 256; if (blk > 8192) blk = 8192;
    hipLaunchKernelGGL(k_f32_to_bf16, dim3((unsigned)blk), dim3(256), 0, stream, s, dptr, n);
  };
  auto gemm = [&](const unsigned short* Aa, int lda, const unsigned short* Bb, int ldb,
                  float* Cc, int ldc, int Mm, int Nn, int Kk, float alpha, int accum,
                  const float* b0, const float* b1, float bs) {
    hipLaunchKernelGGL(gemm_bf16_wmma, dim3(Nn / BN, Mm / BM), dim3(256), 0, stream,
                       Aa, lda, Bb, ldb, Cc, ldc, Mm, Nn, Kk, alpha, accum, b0, b1, bs);
  };

  // ---- 1. convert inputs & weights to bf16 ----
  cvt(x, xbf, (size_t)Mc * Dc);
  cvt(enc_Wr, encWr_bf, (size_t)Ec * Dc);
  cvt(enc_Wi, encWi_bf, (size_t)Ec * Dc);
  for (int p = 0; p < 3; ++p) {
    cvt(pWr[p], pWr_bf[p], (size_t)Hc * Ec * Ec);
    cvt(pWi[p], pWi_bf[p], (size_t)Hc * Ec * Ec);
  }
  cvt(o_Wr, oWr_bf, (size_t)Oc * Hc * Ec);
  cvt(o_Wi, oWi_bf, (size_t)Oc * Hc * Ec);
  cvt(lstmWih, Wih_bf, (size_t)4 * Oc * Oc);

  // ---- 2. encoder: h = x @ (Wr + iWi)^T + (br-bi) + i(br+bi) ----
  gemm(xbf, Dc, encWr_bf, Dc, h_re, Ec, Mc, Ec, Dc, 1.f, 0, enc_br, enc_bi, -1.f);
  gemm(xbf, Dc, encWi_bf, Dc, h_im, Ec, Mc, Ec, Dc, 1.f, 0, enc_br, enc_bi, +1.f);
  cvt(h_re, h_re_bf, (size_t)Mc * Ec);
  cvt(h_im, h_im_bf, (size_t)Mc * Ec);

  // ---- 3. Q/K/V complex head projections ----
  float* preF[3] = { qre, kre, vre };
  float* pimF[3] = { qim, kim, vim };
  for (int p = 0; p < 3; ++p)
    for (int h = 0; h < Hc; ++h) {
      size_t wOff = (size_t)h * Ec * Ec;
      size_t cOff = (size_t)h * Mc * Ec;
      const float* br = pbr[p] + (size_t)h * Ec;
      const float* bi = pbi[p] + (size_t)h * Ec;
      // re = hr@Wr^T - hi@Wi^T + (br-bi)
      gemm(h_re_bf, Ec, pWr_bf[p] + wOff, Ec, preF[p] + cOff, Ec, Mc, Ec, Ec,  1.f, 0, br, bi, -1.f);
      gemm(h_im_bf, Ec, pWi_bf[p] + wOff, Ec, preF[p] + cOff, Ec, Mc, Ec, Ec, -1.f, 1, nullptr, nullptr, 0.f);
      // im = hr@Wi^T + hi@Wr^T + (br+bi)
      gemm(h_re_bf, Ec, pWi_bf[p] + wOff, Ec, pimF[p] + cOff, Ec, Mc, Ec, Ec,  1.f, 0, br, bi, +1.f);
      gemm(h_im_bf, Ec, pWr_bf[p] + wOff, Ec, pimF[p] + cOff, Ec, Mc, Ec, Ec,  1.f, 1, nullptr, nullptr, 0.f);
    }
  cvt(qre, qre_bf, QSZ); cvt(qim, qim_bf, QSZ);
  cvt(kre, kre_bf, QSZ); cvt(kim, kim_bf, QSZ);
  // transpose V (per (h,b) 1024x1024 matrix) so attn@v reuses A@B^T GEMM
  hipLaunchKernelGGL(k_transpose_f32_to_bf16, dim3(Ec / 32, Sc / 32, Hc * Bc),
                     dim3(32, 8), 0, stream, vre, vtre_bf, Sc, Ec);
  hipLaunchKernelGGL(k_transpose_f32_to_bf16, dim3(Ec / 32, Sc / 32, Hc * Bc),
                     dim3(32, 8), 0, stream, vim, vtim_bf, Sc, Ec);

  // ---- 4. attention scores, softmax(|.|), attn @ v ----
  const float scale = 0.03125f;                  // E^-0.5 = 1/32
  for (int h = 0; h < Hc; ++h)
    for (int b = 0; b < Bc; ++b) {
      int z = h * Bc + b;
      size_t qOff = (size_t)z * Sc * Ec;         // == h*M*E + b*S*E
      size_t dOff = (size_t)z * Sc * Sc;
      // dots = q . conj(k):  re = qr@kr^T + qi@ki^T ; im = qi@kr^T - qr@ki^T
      gemm(qre_bf + qOff, Ec, kre_bf + qOff, Ec, dots_re + dOff, Sc, Sc, Sc, Ec,  1.f, 0, nullptr, nullptr, 0.f);
      gemm(qim_bf + qOff, Ec, kim_bf + qOff, Ec, dots_re + dOff, Sc, Sc, Sc, Ec,  1.f, 1, nullptr, nullptr, 0.f);
      gemm(qim_bf + qOff, Ec, kre_bf + qOff, Ec, dots_im + dOff, Sc, Sc, Sc, Ec,  1.f, 0, nullptr, nullptr, 0.f);
      gemm(qre_bf + qOff, Ec, kim_bf + qOff, Ec, dots_im + dOff, Sc, Sc, Sc, Ec, -1.f, 1, nullptr, nullptr, 0.f);
    }
  hipLaunchKernelGGL(k_abs_softmax, dim3(Hc * Bc * Sc), dim3(256), 0, stream,
                     dots_re, dots_im, attn_bf, Sc, scale);
  for (int h = 0; h < Hc; ++h)
    for (int b = 0; b < Bc; ++b) {
      int z = h * Bc + b;
      size_t qOff = (size_t)z * Sc * Ec;
      size_t dOff = (size_t)z * Sc * Sc;
      gemm(attn_bf + dOff, Sc, vtre_bf + qOff, Sc, out_re + qOff, Ec, Sc, Ec, Sc, 1.f, 0, nullptr, nullptr, 0.f);
      gemm(attn_bf + dOff, Sc, vtim_bf + qOff, Sc, out_im + qOff, Ec, Sc, Ec, Sc, 1.f, 0, nullptr, nullptr, 0.f);
    }
  cvt(out_re, out_re_bf, QSZ);
  cvt(out_im, out_im_bf, QSZ);

  // ---- 5. output projection: accumulate per head over K=E slices ----
  for (int h = 0; h < Hc; ++h) {
    size_t aOff = (size_t)h * Mc * Ec;
    const unsigned short* wr = oWr_bf + (size_t)h * Ec;   // [O, H*E] slice, ldb=H*E
    const unsigned short* wi = oWi_bf + (size_t)h * Ec;
    // y_re += cr_h @ oWr_h^T  (bias on first)
    gemm(out_re_bf + aOff, Ec, wr, Hc * Ec, y_re, Oc, Mc, Oc, Ec, 1.f, h > 0,
         h == 0 ? o_br : nullptr, h == 0 ? o_bi : nullptr, -1.f);
    // y_im += cr_h @ oWi_h^T
    gemm(out_re_bf + aOff, Ec, wi, Hc * Ec, y_im, Oc, Mc, Oc, Ec, 1.f, h > 0,
         h == 0 ? o_br : nullptr, h == 0 ? o_bi : nullptr, +1.f);
  }
  for (int h = 0; h < Hc; ++h) {
    size_t aOff = (size_t)h * Mc * Ec;
    const unsigned short* wr = oWr_bf + (size_t)h * Ec;
    const unsigned short* wi = oWi_bf + (size_t)h * Ec;
    gemm(out_im_bf + aOff, Ec, wi, Hc * Ec, y_re, Oc, Mc, Oc, Ec, -1.f, 1, nullptr, nullptr, 0.f);
    gemm(out_im_bf + aOff, Ec, wr, Hc * Ec, y_im, Oc, Mc, Oc, Ec,  1.f, 1, nullptr, nullptr, 0.f);
  }

  // ---- 6. |y| -> layernorm -> n ----
  hipLaunchKernelGGL(k_abs_layernorm, dim3(Mc), dim3(256), 0, stream,
                     y_re, y_im, ln_g, ln_b, n_f, n_bf, Oc);

  // ---- 7. LSTM gates GEMM + gating, final output n + hn ----
  gemm(n_bf, Oc, Wih_bf, Oc, gates, 4 * Oc, Mc, 4 * Oc, Oc, 1.f, 0,
       lstmbih, lstmbhh, +1.f);
  size_t total = (size_t)Mc * Oc;
  hipLaunchKernelGGL(k_lstm_out, dim3((unsigned)((total + 255) / 256)), dim3(256),
                     0, stream, gates, n_f, out, Oc, total);
}
